// NestedNERModel_3435973837039
// MI455X (gfx1250) — compile-verified, matrix operand order checked by reading, not run on hardware
//
#include <hip/hip_runtime.h>
#include <hip/hip_bf16.h>
#include <math.h>

#define B_   64
#define T_   512
#define D_   768
#define H_   384
#define G4_  1536   /* 4*H */
#define L_   33
#define BT_  (B_*T_)
#define NBLK_ 12            /* LSTM blocks per direction */
#define CPB_  32            /* h-columns per LSTM block (H/NBLK) */

typedef __attribute__((ext_vector_type(16))) _Float16 v16h;
typedef __attribute__((ext_vector_type(8)))  float    v8f;
typedef unsigned int u32x4 __attribute__((ext_vector_type(4)));
typedef int          i32x4 __attribute__((ext_vector_type(4)));
typedef int          i32x8 __attribute__((ext_vector_type(8)));

#ifndef __has_builtin
#define __has_builtin(x) 0
#endif
#if __has_builtin(__builtin_amdgcn_tensor_load_to_lds) && \
    __has_builtin(__builtin_amdgcn_s_wait_tensorcnt)
#define USE_TDM 1
#else
#define USE_TDM 0
#endif

// ---- WMMA helpers (CDNA5 wave32 layouts, CDNA5 ISA) ----
__device__ __forceinline__ v16h ld_frag(const _Float16* base, int ld, int lane) {
  int row = lane & 15;
  int kb  = (lane >> 4) << 3;
  const _Float16* p = base + row * ld + kb;
  v16h f;
  ((float4*)&f)[0] = *(const float4*)(p);
  ((float4*)&f)[1] = *(const float4*)(p + 16);
  return f;
}

__device__ __forceinline__ v8f wmma16(v16h a, v16h b, v8f c) {
  return __builtin_amdgcn_wmma_f32_16x16x32_f16(false, a, false, b, (short)0, c,
                                                false, false);
}

__device__ __forceinline__ float sigm(float x) { return 1.0f / (1.0f + __expf(-x)); }

#if USE_TDM
// Tensor Data Mover: LDS <- global tile load. Descriptor per
// CDNA5 ISA. data_size = 2 bytes (f16).
// 3D tile: tile_d0 contiguous elements per line, tile_d1 lines (stride
// d0_stride elems), tile_d2 planes (stride d1_stride elems).
__device__ __forceinline__ void tdm_load_f16(unsigned lds_off,
                                             unsigned long long gaddr,
                                             unsigned tensor_d0, unsigned tensor_d1,
                                             unsigned tensor_d2,
                                             unsigned tile_d0, unsigned tile_d1,
                                             unsigned tile_d2,
                                             unsigned long long d0_stride,
                                             unsigned long long d1_stride) {
  u32x4 g0;
  g0[0] = 1u;                                               // count=1, user D#
  g0[1] = lds_off;                                          // lds_addr (bytes)
  g0[2] = (unsigned)(gaddr & 0xFFFFFFFFu);                  // global_addr[31:0]
  g0[3] = (unsigned)((gaddr >> 32) & 0x1FFFFFFu) | (2u << 30);  // [56:32]|type=2
  i32x8 g1;
  g1[0] = (int)(1u << 16);                                  // data_size=2B
  g1[1] = (int)((tensor_d0 & 0xFFFFu) << 16);               // dim0[15:0]
  g1[2] = (int)(((tensor_d0 >> 16) & 0xFFFFu) |             // dim0[31:16]
                ((tensor_d1 & 0xFFFFu) << 16));             // dim1[15:0]
  g1[3] = (int)(((tensor_d1 >> 16) & 0xFFFFu) |             // dim1[31:16]
                ((tile_d0 & 0xFFFFu) << 16));               // tile_dim0
  g1[4] = (int)((tile_d1 & 0xFFFFu) | ((tile_d2 & 0xFFFFu) << 16));
  g1[5] = (int)(unsigned)(d0_stride & 0xFFFFFFFFu);         // dim0_stride[31:0]
  g1[6] = (int)((unsigned)((d0_stride >> 32) & 0xFFFFu) |   // dim0_stride[47:32]
                (unsigned)((d1_stride & 0xFFFFu) << 16));   // dim1_stride[15:0]
  g1[7] = (int)(unsigned)((d1_stride >> 16) & 0xFFFFFFFFu); // dim1_stride[47:16]
  i32x4 g2 = {(int)tensor_d2, 0, 0, 0};
  i32x4 g3 = {0, 0, 0, 0};
#if defined(__clang_major__) && (__clang_major__ >= 23)
  i32x8 g4 = {0, 0, 0, 0, 0, 0, 0, 0};
  __builtin_amdgcn_tensor_load_to_lds(g0, g1, g2, g3, g4, 0);
#else
  __builtin_amdgcn_tensor_load_to_lds(g0, g1, g2, g3, 0);
#endif
}
#endif

// ---------------- small prep kernels ----------------
__global__ void k_init(const unsigned char* mask, int* len, int* cnt) {
  int b = threadIdx.x;
  if (b < 2) cnt[b] = 0;  // grid-barrier counters (re-zeroed every launch)
  if (b < B_) {
    int c = 0;
    for (int t = 0; t < T_; ++t) c += mask[b * T_ + t] ? 1 : 0;
    len[b] = c;
  }
}

__global__ void k_f2h(const float* s, _Float16* d, int n) {
  int i = blockIdx.x * blockDim.x + threadIdx.x;
  if (i < n) d[i] = (_Float16)s[i];
}

__global__ void k_padlin(const float* s, _Float16* d) {
  int i = blockIdx.x * blockDim.x + threadIdx.x;
  if (i < 48 * D_) {
    int r = i / D_;
    d[i] = (r < L_) ? (_Float16)s[i] : (_Float16)0.0f;
  }
}

__global__ void k_xrev(const float* x, const int* len, _Float16* x16, _Float16* xr16) {
  int i = blockIdx.x * blockDim.x + threadIdx.x;
  if (i < B_ * T_ * D_) {
    int d  = i % D_;
    int bt = i / D_;
    int t  = bt % T_;
    int b  = bt / T_;
    x16[i] = (_Float16)x[i];
    int rt = len[b] - 1 - t;
    rt = rt < 0 ? 0 : (rt > T_ - 1 ? T_ - 1 : rt);
    xr16[i] = (_Float16)x[(b * T_ + rt) * D_ + d];
  }
}

// ---------------- input projection GEMM (WMMA) ----------------
__global__ void __launch_bounds__(256) k_gemm_xp(const _Float16* __restrict__ X,
                                                 const _Float16* __restrict__ W,
                                                 const float* __restrict__ bias,
                                                 float* __restrict__ out) {
  __shared__ _Float16 As[64 * 32];
  int wave = threadIdx.x >> 5, lane = threadIdx.x & 31;
  int m_sub = wave & 3, n_half = wave >> 2;
  v8f acc[4] = {{}, {}, {}, {}};
  int mbase = blockIdx.x * 64;
  int nbase = blockIdx.y * 128 + n_half * 64;
  int srow = threadIdx.x >> 2;
  int scol = (threadIdx.x & 3) << 3;

  for (int kt = 0; kt < 24; ++kt) {
    int k0 = kt * 32;
    __syncthreads();
    *(float4*)(&As[srow * 32 + scol]) =
        *(const float4*)(X + (mbase + srow) * D_ + k0 + scol);
    __syncthreads();
    v16h a = ld_frag(&As[m_sub * 16 * 32], 32, lane);
    const _Float16* wp = W + nbase * D_ + k0;
    __builtin_prefetch(wp + 32, 0, 1);  // global_prefetch_b8
#pragma unroll
    for (int nt = 0; nt < 4; ++nt) {
      v16h b = ld_frag(wp + nt * 16 * D_, D_, lane);
      acc[nt] = wmma16(a, b, acc[nt]);
    }
  }
  int nlane = lane & 15, moff = (lane >> 4) << 3;
#pragma unroll
  for (int nt = 0; nt < 4; ++nt) {
    int col = nbase + nt * 16 + nlane;
    float bv = bias[col];
#pragma unroll
    for (int r = 0; r < 8; ++r) {
      int row = mbase + m_sub * 16 + r + moff;
      out[row * G4_ + col] = acc[nt][r] + bv;
    }
  }
}

// ---------------- persistent multi-WGP LSTM recurrence ----------------
// grid = (NBLK_, 2). Each block owns CPB_=32 h-columns; its 4-gate weight
// slice (128x384 f16, 96KB) is TDM-loaded once into LDS and stays resident.
// The h mirror (64x384 f16, 48KB) is refreshed each step with one TDM
// descriptor (s_wait_tensorcnt), after an atomic grid barrier.
__global__ void __launch_bounds__(256) k_lstm(const float* __restrict__ xp_f,
                                              const float* __restrict__ xp_b,
                                              const _Float16* __restrict__ whh_f,
                                              const _Float16* __restrict__ whh_b,
                                              const int* __restrict__ len,
                                              _Float16* __restrict__ hGf,
                                              _Float16* __restrict__ hGb,
                                              _Float16* __restrict__ hf_out,
                                              _Float16* __restrict__ hb_out,
                                              int* __restrict__ cnt) {
  extern __shared__ char smem[];
  _Float16* wlds = (_Float16*)smem;                    // [128][384] = 96KB
  _Float16* hbuf = (_Float16*)(smem + 128 * H_ * 2);   // [64][384]  = 48KB

  int blkn = blockIdx.x;
  int dir  = blockIdx.y;
  const float*    xp  = dir ? xp_b : xp_f;
  const _Float16* whh = dir ? whh_b : whh_f;
  _Float16* hG   = dir ? hGb : hGf;
  _Float16* hout = dir ? hb_out : hf_out;
  int* cnt_d = cnt + dir;
  int tid = threadIdx.x;

  // ---- stage weight slice (rows g*32+lj <-> global g*384 + blkn*32 + lj) ----
#if USE_TDM
  unsigned wlds_off = (unsigned)(size_t)(void*)wlds;
  unsigned hbuf_off = (unsigned)(size_t)(void*)hbuf;
  if (tid < 32) {
    // 3D tile: 384 elems/line, 32 lines (stride 384), 4 planes (stride 384*384)
    tdm_load_f16(wlds_off, (unsigned long long)(size_t)(whh + blkn * CPB_ * H_),
                 /*tensor*/ H_, CPB_, 4, /*tile*/ H_, CPB_, 4,
                 /*strides*/ H_, (unsigned long long)H_ * H_);
    __builtin_amdgcn_s_wait_tensorcnt(0);
  }
#else
  for (int u = tid; u < 128 * H_ / 8; u += 256) {
    int h8 = u * 8;
    int lrow = h8 / H_, lcol = h8 % H_;
    int grow = (lrow >> 5) * H_ + blkn * CPB_ + (lrow & 31);
    *(float4*)(wlds + lrow * H_ + lcol) = *(const float4*)(whh + grow * H_ + lcol);
  }
#endif
  // zero h mirror
  float4 z4 = {0.f, 0.f, 0.f, 0.f};
  for (int u = tid; u < 64 * H_ / 8; u += 256) ((float4*)hbuf)[u] = z4;
  __syncthreads();

  int wave = tid >> 5, lane = tid & 31;
  int nlane = lane & 15, moff = (lane >> 4) << 3;
  int m   = wave & 3;        // batch tile 0..3
  int jtl = wave >> 2;       // local 16-col tile 0..1
  int jloc = jtl * 16 + nlane;
  int jglob = blkn * CPB_ + jloc;

  float cs[8];
  int mylen[8];
#pragma unroll
  for (int r = 0; r < 8; ++r) {
    cs[r] = 0.0f;
    mylen[r] = len[m * 16 + r + moff];
  }

  for (int t = 0; t < T_; ++t) {
    v8f acc[4] = {{}, {}, {}, {}};
#pragma unroll 1
    for (int kt = 0; kt < 12; ++kt) {
      int k0 = kt * 32;
      v16h a = ld_frag(hbuf + m * 16 * H_ + k0, H_, lane);  // ds_load_b128
#pragma unroll
      for (int g = 0; g < 4; ++g) {
        v16h b = ld_frag(wlds + (g * CPB_ + jtl * 16) * H_ + k0, H_, lane);
        acc[g] = wmma16(a, b, acc[g]);
      }
    }
    // gate nonlinearity + state update for this block's 32-column slice
#pragma unroll
    for (int r = 0; r < 8; ++r) {
      int brow = m * 16 + r + moff;
      const float* xr = xp + (size_t)(brow * T_ + t) * G4_;
      float xi = acc[0][r] + xr[jglob];
      float xf = acc[1][r] + xr[H_ + jglob];
      float xg = acc[2][r] + xr[2 * H_ + jglob];
      float xo = acc[3][r] + xr[3 * H_ + jglob];
      float cn = sigm(xf) * cs[r] + sigm(xi) * tanhf(xg);
      float hn = sigm(xo) * tanhf(cn);
      bool mk = (t < mylen[r]);
      float hold = (float)hbuf[brow * H_ + jglob];
      cs[r] = mk ? cn : cs[r];
      hG[brow * H_ + jglob] = (_Float16)(mk ? hn : hold);
      hout[(size_t)(brow * T_ + t) * H_ + jglob] = (_Float16)(mk ? hn : 0.0f);
    }
    __threadfence();   // h slice visible device-scope
    __syncthreads();
    if (tid == 0) {    // grid barrier (monotonic counter, per direction)
      atomicAdd(cnt_d, 1);
      int target = NBLK_ * (t + 1);
      while (__hip_atomic_load(cnt_d, __ATOMIC_ACQUIRE, __HIP_MEMORY_SCOPE_AGENT) <
             target) {
        __builtin_amdgcn_s_sleep(2);
      }
    }
    __syncthreads();
    // refresh full h mirror from L2
#if USE_TDM
    if (tid < 32) {
      tdm_load_f16(hbuf_off, (unsigned long long)(size_t)hG,
                   /*tensor*/ 64 * H_, 1, 0, /*tile*/ 64 * H_ / 4, 4, 0,
                   /*strides*/ 64 * H_ / 4, 0);
      __builtin_amdgcn_s_wait_tensorcnt(0);
    }
#else
    for (int u = tid; u < 64 * H_ / 8; u += 256)
      ((float4*)hbuf)[u] = ((const float4*)hG)[u];
#endif
    __syncthreads();
  }
}

// ---------------- concat fwd + un-reversed bwd ----------------
__global__ void k_cat(const _Float16* hf, const _Float16* hb, const int* len,
                      _Float16* hcat) {
  int i = blockIdx.x * blockDim.x + threadIdx.x;
  if (i < B_ * T_ * H_) {
    int j = i % H_;
    int bt = i / H_;
    int t = bt % T_;
    int b = bt / T_;
    hcat[bt * (2 * H_) + j] = hf[i];
    int lb = len[b];
    int rt = lb - 1 - t;
    rt = rt < 0 ? 0 : (rt > T_ - 1 ? T_ - 1 : rt);
    hcat[bt * (2 * H_) + H_ + j] =
        (t < lb) ? hb[(b * T_ + rt) * H_ + j] : (_Float16)0.0f;
  }
}

// ---------------- classifier head (WMMA, N padded 33->48) ----------------
__global__ void __launch_bounds__(256) k_linear(const _Float16* __restrict__ hcat,
                                                const _Float16* __restrict__ wlin,
                                                const float* __restrict__ lin_b,
                                                const int* __restrict__ len,
                                                float* __restrict__ em) {
  int wave = threadIdx.x >> 5, lane = threadIdx.x & 31;
  int id = blockIdx.x * 8 + wave;  // 6144 tiles = 2048(M) * 3(N)
  int mtile = id / 3, ntile = id % 3;
  v8f acc = {};
  const _Float16* arow = hcat + mtile * 16 * (2 * H_);
  const _Float16* brow = wlin + ntile * 16 * (2 * H_);
  for (int kt = 0; kt < 24; ++kt) {
    v16h a = ld_frag(arow + kt * 32, 2 * H_, lane);
    v16h b = ld_frag(brow + kt * 32, 2 * H_, lane);
    acc = wmma16(a, b, acc);
  }
  int col = ntile * 16 + (lane & 15);
  int moff = (lane >> 4) << 3;
  if (col < L_) {
    float bv = lin_b[col];
#pragma unroll
    for (int r = 0; r < 8; ++r) {
      int row = mtile * 16 + r + moff;
      int t = row % T_, b = row / T_;
      float mk = (t < len[b]) ? 1.0f : 0.0f;
      em[row * L_ + col] = (acc[r] + bv) * mk;
    }
  }
}

// ---------------- CRF numerator ----------------
__global__ void k_num(const int* labels, const float* trans, const float* st,
                      const float* et, const float* em, const int* len,
                      float* numb) {
  int b = threadIdx.x;
  if (b >= B_) return;
  const int* lab = labels + b * T_;
  int lb = len[b];
  float num = st[lab[0]] + em[(b * T_) * L_ + lab[0]];
  for (int t = 1; t < T_; ++t)
    if (t < lb) num += trans[lab[t - 1] * L_ + lab[t]] + em[(b * T_ + t) * L_ + lab[t]];
  num += et[lab[lb - 1]];
  numb[b] = num;
}

// ---------------- CRF forward (logZ) ----------------
__global__ void k_alpha(const float* trans, const float* st, const float* et,
                        const float* em, const int* len, const float* numb,
                        float* score) {
  __shared__ float al[L_], nb[L_];
  int j = threadIdx.x;
  int b = blockIdx.x;
  const float* emb = em + (size_t)b * T_ * L_;
  int lb = len[b];
  if (j < L_) al[j] = st[j] + emb[j];
  __syncthreads();
  for (int t = 1; t < T_; ++t) {
    if (t < lb) {  // uniform within block
      if (j < L_) {
        float mx = -1e30f;
        for (int i = 0; i < L_; ++i) mx = fmaxf(mx, al[i] + trans[i * L_ + j]);
        float s = 0.0f;
        for (int i = 0; i < L_; ++i) s += __expf(al[i] + trans[i * L_ + j] - mx);
        nb[j] = mx + __logf(s) + emb[t * L_ + j];
      }
      __syncthreads();
      if (j < L_) al[j] = nb[j];
      __syncthreads();
    }
  }
  if (j == 0) {
    float mx = -1e30f;
    for (int i = 0; i < L_; ++i) mx = fmaxf(mx, al[i] + et[i]);
    float s = 0.0f;
    for (int i = 0; i < L_; ++i) s += __expf(al[i] + et[i] - mx);
    score[b] = numb[b] - (mx + __logf(s));
  }
}

__global__ void k_reduce(const float* score, float* out) {
  __shared__ float sh[64];
  int t = threadIdx.x;
  sh[t] = score[t];
  __syncthreads();
  for (int s = 32; s > 0; s >>= 1) {
    if (t < s) sh[t] += sh[t + s];
    __syncthreads();
  }
  if (t == 0) out[0] = -sh[0] / (float)B_;
}

// ---------------- host orchestration ----------------
extern "C" void kernel_launch(void* const* d_in, const int* in_sizes, int n_in,
                              void* d_out, int out_size, void* d_ws, size_t ws_size,
                              hipStream_t stream) {
  (void)in_sizes; (void)n_in; (void)out_size; (void)ws_size;
  const float*         x      = (const float*)d_in[0];
  const unsigned char* mask   = (const unsigned char*)d_in[1];
  const int*           labels = (const int*)d_in[2];
  /* d_in[3] (index) unused by the reference */
  const float* w_ih_f = (const float*)d_in[4];
  const float* w_hh_f = (const float*)d_in[5];
  const float* b_f    = (const float*)d_in[6];
  const float* w_ih_b = (const float*)d_in[7];
  const float* w_hh_b = (const float*)d_in[8];
  const float* b_b    = (const float*)d_in[9];
  const float* lin_w  = (const float*)d_in[10];
  const float* lin_b  = (const float*)d_in[11];
  const float* trans  = (const float*)d_in[12];
  const float* st     = (const float*)d_in[13];
  const float* et     = (const float*)d_in[14];
  float* out = (float*)d_out;

  char* ws = (char*)d_ws;
  size_t off = 0;
  auto take = [&](size_t bytes) {
    size_t r = off;
    off += (bytes + 255) & ~(size_t)255;
    return r;
  };
  _Float16* X16   = (_Float16*)(ws + take((size_t)B_ * T_ * D_ * 2));
  _Float16* XR16  = (_Float16*)(ws + take((size_t)B_ * T_ * D_ * 2));
  _Float16* WIHF  = (_Float16*)(ws + take((size_t)G4_ * D_ * 2));
  _Float16* WIHB  = (_Float16*)(ws + take((size_t)G4_ * D_ * 2));
  _Float16* WHHF  = (_Float16*)(ws + take((size_t)G4_ * H_ * 2));
  _Float16* WHHB  = (_Float16*)(ws + take((size_t)G4_ * H_ * 2));
  _Float16* LINW  = (_Float16*)(ws + take((size_t)48 * D_ * 2));
  float*    XPF   = (float*)(ws + take((size_t)BT_ * G4_ * 4));
  float*    XPB   = (float*)(ws + take((size_t)BT_ * G4_ * 4));
  _Float16* HF    = (_Float16*)(ws + take((size_t)BT_ * H_ * 2));
  _Float16* HB    = (_Float16*)(ws + take((size_t)BT_ * H_ * 2));
  _Float16* HCAT  = (_Float16*)(ws + take((size_t)BT_ * 2 * H_ * 2));
  float*    EM    = (float*)(ws + take((size_t)BT_ * L_ * 4));
  _Float16* HGF   = (_Float16*)(ws + take((size_t)B_ * H_ * 2));
  _Float16* HGB   = (_Float16*)(ws + take((size_t)B_ * H_ * 2));
  int*      LEN   = (int*)(ws + take((size_t)B_ * 4));
  int*      CNT   = (int*)(ws + take((size_t)2 * 4));
  float*    NUMB  = (float*)(ws + take((size_t)B_ * 4));
  float*    SCORE = (float*)(ws + take((size_t)B_ * 4));

  const int lstm_smem = (128 * H_ + 64 * H_) * 2;  // 144KB dynamic LDS
  (void)hipFuncSetAttribute((const void*)k_lstm,
                            hipFuncAttributeMaxDynamicSharedMemorySize, lstm_smem);

  k_init<<<1, 64, 0, stream>>>(mask, LEN, CNT);

  int n;
  n = G4_ * D_; k_f2h<<<(n + 255) / 256, 256, 0, stream>>>(w_ih_f, WIHF, n);
  n = G4_ * D_; k_f2h<<<(n + 255) / 256, 256, 0, stream>>>(w_ih_b, WIHB, n);
  n = G4_ * H_; k_f2h<<<(n + 255) / 256, 256, 0, stream>>>(w_hh_f, WHHF, n);
  n = G4_ * H_; k_f2h<<<(n + 255) / 256, 256, 0, stream>>>(w_hh_b, WHHB, n);
  n = 48 * D_;  k_padlin<<<(n + 255) / 256, 256, 0, stream>>>(lin_w, LINW);

  n = B_ * T_ * D_;
  k_xrev<<<(n + 255) / 256, 256, 0, stream>>>(x, LEN, X16, XR16);

  dim3 gg(BT_ / 64, G4_ / 128);  // 512 x 12
  k_gemm_xp<<<gg, 256, 0, stream>>>(X16, WIHF, b_f, XPF);
  k_gemm_xp<<<gg, 256, 0, stream>>>(XR16, WIHB, b_b, XPB);

  dim3 gl(NBLK_, 2);
  k_lstm<<<gl, 256, lstm_smem, stream>>>(XPF, XPB, WHHF, WHHB, LEN, HGF, HGB,
                                         HF, HB, CNT);

  n = B_ * T_ * H_;
  k_cat<<<(n + 255) / 256, 256, 0, stream>>>(HF, HB, LEN, HCAT);

  k_linear<<<768, 256, 0, stream>>>(HCAT, LINW, lin_b, LEN, EM);

  k_num<<<1, 64, 0, stream>>>(labels, trans, st, et, EM, LEN, NUMB);
  k_alpha<<<64, 64, 0, stream>>>(trans, st, et, EM, LEN, NUMB, SCORE);
  k_reduce<<<1, 64, 0, stream>>>(SCORE, out);
}
